// GroupedKNNEstimator_19396072309095
// MI455X (gfx1250) — matrix operand order repaired and checked by reading, not compiled, not run on hardware
//
#include <hip/hip_runtime.h>
#include <hip/hip_bf16.h>
#include <math.h>

// CDNA5 wave32 WMMA vector types
typedef __attribute__((ext_vector_type(16))) __bf16 v16bf;
typedef __attribute__((ext_vector_type(8)))  float  v8f;

namespace {
constexpr int kN = 2048;          // queries
constexpr int kM = 100000;        // memory bank rows
constexpr int kD = 128;           // feature dim
constexpr int kKnn = 5;
constexpr int kChunks = 125;      // M split into 125 chunks of 800 rows
constexpr int kTilesPerChunk = 50;// 50 * 16 = 800
constexpr int kQTiles = kN / 16;  // 128
constexpr int kPitch = 17;        // LDS row pitch (conflict padding)
}

__device__ __forceinline__ void insert5(float d, float& t0, float& t1,
                                        float& t2, float& t3, float& t4) {
  if (d < t4) {
    if (d < t0)      { t4 = t3; t3 = t2; t2 = t1; t1 = t0; t0 = d; }
    else if (d < t1) { t4 = t3; t3 = t2; t2 = t1; t1 = d; }
    else if (d < t2) { t4 = t3; t3 = t2; t2 = d; }
    else if (d < t3) { t4 = t3; t3 = d; }
    else             { t4 = d; }
  }
}

// Split f32 rows into bf16 hi/lo and compute per-row sum of squares.
// One block per row, 128 threads (= D).
__global__ __launch_bounds__(128)
void split_bf16_kernel(const float* __restrict__ src,
                       __bf16* __restrict__ hi, __bf16* __restrict__ lo,
                       float* __restrict__ sq) {
  __shared__ float red[128];
  const int row = blockIdx.x;
  const int t   = threadIdx.x;
  const float x = src[(size_t)row * kD + t];
  const __bf16 h = (__bf16)x;           // RNE
  const float  hv = (float)h;
  const __bf16 l = (__bf16)(x - hv);    // residual
  hi[(size_t)row * kD + t] = h;
  lo[(size_t)row * kD + t] = l;
  red[t] = x * x;
  __syncthreads();
  for (int s = 64; s > 0; s >>= 1) {
    if (t < s) red[t] += red[t + s];
    __syncthreads();
  }
  if (t == 0) sq[row] = red[0];
}

// One wave (32 threads) per (query-tile, bank-chunk).
// GEMM via split-bf16 WMMA with 3 independent accumulator chains;
// per-lane running top-5 of squared distances.
__global__ __launch_bounds__(32)
void knn_chunk_kernel(const __bf16* __restrict__ Ahi, const __bf16* __restrict__ Alo,
                      const float* __restrict__ x2,
                      const __bf16* __restrict__ Bhi, const __bf16* __restrict__ Blo,
                      const float* __restrict__ y2,
                      float* __restrict__ partials) {
  __shared__ float lds[16 * kPitch];
  const int lane = threadIdx.x;
  const int idx  = lane & 15;   // row (A) / col (B) / col (C) within tile
  const int half = lane >> 4;   // selects K sub-range per ISA layout
  const int qtile = blockIdx.x;
  const int chunk = blockIdx.y;

  union V16 { v16bf v; uint4 u[2]; };

  // ---- A tile (16 queries x 128), kept in registers for the whole chunk ----
  V16 ahi[4], alo[4];
  {
    const size_t qrow = (size_t)(qtile * 16 + idx);
    const __bf16* pAhi = Ahi + qrow * kD;
    const __bf16* pAlo = Alo + qrow * kD;
#pragma unroll
    for (int k = 0; k < 4; ++k) {
      const int kb = k * 32;
      // elements 0..7 : K = kb + 8*half + {0..7}; elements 8..15: +16
      ahi[k].u[0] = *(const uint4*)(pAhi + kb + 8 * half);
      ahi[k].u[1] = *(const uint4*)(pAhi + kb + 16 + 8 * half);
      alo[k].u[0] = *(const uint4*)(pAlo + kb + 8 * half);
      alo[k].u[1] = *(const uint4*)(pAlo + kb + 16 + 8 * half);
    }
  }
  // |x|^2 for the 8 C rows (m = 8*half + v) this lane accumulates
  float x2q[8];
  {
    const float4 a = *(const float4*)(x2 + qtile * 16 + 8 * half);
    const float4 b = *(const float4*)(x2 + qtile * 16 + 8 * half + 4);
    x2q[0] = a.x; x2q[1] = a.y; x2q[2] = a.z; x2q[3] = a.w;
    x2q[4] = b.x; x2q[5] = b.y; x2q[6] = b.z; x2q[7] = b.w;
  }

  float t0 = 3.0e38f, t1 = 3.0e38f, t2 = 3.0e38f, t3 = 3.0e38f, t4 = 3.0e38f;
  const int bank0 = chunk * (kTilesPerChunk * 16);

  for (int bt = 0; bt < kTilesPerChunk; ++bt) {
    const int bankrow = bank0 + bt * 16 + idx;
    const __bf16* pBhi = Bhi + (size_t)bankrow * kD;
    const __bf16* pBlo = Blo + (size_t)bankrow * kD;
    const float y2n = y2[bankrow];

    // three independent accumulator chains (4-deep each instead of 12-deep)
    v8f c0 = {};  // hi * hi
    v8f c1 = {};  // hi * lo
    v8f c2 = {};  // lo * hi
#pragma unroll
    for (int k = 0; k < 4; ++k) {
      V16 bh, bl;
      const int kb = k * 32 + 8 * half;  // elements 0..7 : K = kb + {0..7}; elements 8..15: +16
      bh.u[0] = *(const uint4*)(pBhi + kb);
      bh.u[1] = *(const uint4*)(pBhi + kb + 16);
      bl.u[0] = *(const uint4*)(pBlo + kb);
      bl.u[1] = *(const uint4*)(pBlo + kb + 16);
      c0 = __builtin_amdgcn_wmma_f32_16x16x32_bf16(false, ahi[k].v, false, bh.v,
                                                   (short)0, c0, false, false);
      c1 = __builtin_amdgcn_wmma_f32_16x16x32_bf16(false, ahi[k].v, false, bl.v,
                                                   (short)0, c1, false, false);
      c2 = __builtin_amdgcn_wmma_f32_16x16x32_bf16(false, alo[k].v, false, bh.v,
                                                   (short)0, c2, false, false);
    }
    if (bt + 1 < kTilesPerChunk)
      __builtin_prefetch(pBhi + (size_t)16 * kD, 0, 1);  // next 16-row B tile

    // d^2 = |x|^2 + |y|^2 - 2 x.y  ->  LDS[m][n]
#pragma unroll
    for (int v = 0; v < 8; ++v) {
      const float dot = c0[v] + c1[v] + c2[v];
      const float d2  = x2q[v] + y2n - 2.0f * dot;
      lds[(8 * half + v) * kPitch + idx] = d2;
    }
    __syncthreads();
    // lane (idx, half) scans its half of row idx's 16 columns
    {
      const float* rowp = &lds[idx * kPitch + 8 * half];
#pragma unroll
      for (int j = 0; j < 8; ++j) insert5(rowp[j], t0, t1, t2, t3, t4);
    }
    __syncthreads();
  }

  // merge the two half-lane top-5 sets per query row
  if (half == 1) {
    lds[idx * kPitch + 0] = t0; lds[idx * kPitch + 1] = t1;
    lds[idx * kPitch + 2] = t2; lds[idx * kPitch + 3] = t3;
    lds[idx * kPitch + 4] = t4;
  }
  __syncthreads();
  if (half == 0) {
#pragma unroll
    for (int j = 0; j < 5; ++j) insert5(lds[idx * kPitch + j], t0, t1, t2, t3, t4);
    float* dst = partials + ((size_t)(qtile * 16 + idx) * kChunks + chunk) * kKnn;
    dst[0] = t0; dst[1] = t1; dst[2] = t2; dst[3] = t3; dst[4] = t4;
  }
}

// Merge per-chunk candidates, apply sqrt + min-max normalization, mean.
__global__ __launch_bounds__(256)
void knn_finalize_kernel(const float* __restrict__ partials,
                         const float* __restrict__ minv_p,
                         const float* __restrict__ maxv_p,
                         float* __restrict__ out) {
  const int q = blockIdx.x * blockDim.x + threadIdx.x;
  if (q >= kN) return;
  const float* p = partials + (size_t)q * kChunks * kKnn;
  float t0 = 3.0e38f, t1 = 3.0e38f, t2 = 3.0e38f, t3 = 3.0e38f, t4 = 3.0e38f;
  for (int i = 0; i < kChunks * kKnn; ++i) insert5(p[i], t0, t1, t2, t3, t4);
  const float minv = *minv_p;
  const float inv = 1.0f / (*maxv_p - minv);
  float s = 0.0f;
  s += (sqrtf(fmaxf(t0, 0.0f)) - minv) * inv;
  s += (sqrtf(fmaxf(t1, 0.0f)) - minv) * inv;
  s += (sqrtf(fmaxf(t2, 0.0f)) - minv) * inv;
  s += (sqrtf(fmaxf(t3, 0.0f)) - minv) * inv;
  s += (sqrtf(fmaxf(t4, 0.0f)) - minv) * inv;
  out[q] = s * 0.2f;
}

extern "C" void kernel_launch(void* const* d_in, const int* in_sizes, int n_in,
                              void* d_out, int out_size, void* d_ws, size_t ws_size,
                              hipStream_t stream) {
  const float* features = (const float*)d_in[0];   // (2048,128)
  const float* bank     = (const float*)d_in[1];   // (100000,128)
  const float* minv     = (const float*)d_in[2];   // scalar
  const float* maxv     = (const float*)d_in[3];   // scalar
  (void)in_sizes; (void)n_in; (void)out_size;      // n_neighbors==5 baked in

  // workspace layout (~58 MB): bf16 hi/lo copies + row norms + partial top-5
  char* ws = (char*)d_ws;
  size_t off = 0;
  auto align16 = [](size_t o) { return (o + 15) & ~(size_t)15; };
  __bf16* Bhi = (__bf16*)(ws + off); off += (size_t)kM * kD * sizeof(__bf16);
  __bf16* Blo = (__bf16*)(ws + off); off += (size_t)kM * kD * sizeof(__bf16);
  float*  y2  = (float*)(ws + off);  off += (size_t)kM * sizeof(float);
  off = align16(off);
  __bf16* Ahi = (__bf16*)(ws + off); off += (size_t)kN * kD * sizeof(__bf16);
  __bf16* Alo = (__bf16*)(ws + off); off += (size_t)kN * kD * sizeof(__bf16);
  float*  x2  = (float*)(ws + off);  off += (size_t)kN * sizeof(float);
  off = align16(off);
  float* partials = (float*)(ws + off);
  (void)ws_size;  // requires ~58 MB of scratch

  float* out = (float*)d_out;

  split_bf16_kernel<<<kM, 128, 0, stream>>>(bank, Bhi, Blo, y2);
  split_bf16_kernel<<<kN, 128, 0, stream>>>(features, Ahi, Alo, x2);
  knn_chunk_kernel<<<dim3(kQTiles, kChunks), 32, 0, stream>>>(
      Ahi, Alo, x2, Bhi, Blo, y2, partials);
  knn_finalize_kernel<<<(kN + 255) / 256, 256, 0, stream>>>(
      partials, minv, maxv, out);
}
